// GRU_23613730193671
// MI455X (gfx1250) — matrix operand order repaired and checked by reading, not compiled, not run on hardware
//
#include <hip/hip_runtime.h>
#include <hip/hip_bf16.h>

typedef __attribute__((ext_vector_type(16))) __bf16    v16bf;
typedef __attribute__((ext_vector_type(8)))  float     v8f;
typedef __attribute__((ext_vector_type(4)))  unsigned  v4u;

union FragBF { v16bf v; v4u q[2]; };

__device__ __forceinline__ unsigned short f2bf(float f) {
    unsigned x = __float_as_uint(f);
    unsigned r = x + 0x7FFFu + ((x >> 16) & 1u);   // round-to-nearest-even
    return (unsigned short)(r >> 16);
}
__device__ __forceinline__ float bf2f(unsigned short h) {
    return __uint_as_float(((unsigned)h) << 16);
}

// A fragment: 16x32 bf16 row-major; lane r=l&15 holds row r,
// K = {half*8 .. half*8+7} and {16+half*8 .. 16+half*8+7}
__device__ __forceinline__ v16bf load_a_frag(const unsigned short* __restrict__ base,
                                             int ld, int lane) {
    int r = lane & 15, half = lane >> 4;
    const unsigned short* p = base + (size_t)r * ld + half * 8;
    FragBF f;
    f.q[0] = *(const v4u*)(p);
    f.q[1] = *(const v4u*)(p + 16);
    return f.v;
}

// B fragment: B = W^T (32x16); W stored [N,K] row-major -> lane loads W row
// (n0 + (lane&15)), K = {half*8 .. half*8+7} and {16+half*8 .. 16+half*8+7}
__device__ __forceinline__ v16bf load_b_frag(const unsigned short* __restrict__ wrow_base,
                                             int ld, int lane) {
    int r = lane & 15, half = lane >> 4;
    const unsigned short* p = wrow_base + (size_t)r * ld + half * 8;
    FragBF f;
    f.q[0] = *(const v4u*)(p);
    f.q[1] = *(const v4u*)(p + 16);
    return f.v;
}

__device__ __forceinline__ v8f wmma_bf16(v16bf a, v16bf b, v8f c) {
    return __builtin_amdgcn_wmma_f32_16x16x32_bf16(false, a, false, b, (short)0, c,
                                                   false, false);
}

// ---------------- utility kernels ----------------

__global__ void k_f32_to_bf16(const float* __restrict__ src,
                              unsigned short* __restrict__ dst, int n) {
    int i = blockIdx.x * blockDim.x + threadIdx.x;
    if (i < n) dst[i] = f2bf(src[i]);
}

__global__ void k_zero_bf16(unsigned short* __restrict__ dst, int n) {
    int i = blockIdx.x * blockDim.x + threadIdx.x;
    if (i < n) dst[i] = 0;
}

__global__ void k_zero_u32(unsigned* __restrict__ dst, int n) {
    int i = blockIdx.x * blockDim.x + threadIdx.x;
    if (i < n) dst[i] = 0u;
}

// ---------------- GEMM: out[M,N] = A[M,K](bf16) @ W[N,K]^T(bf16) + bias[N] ----------------
// Wave tile: 32(M) x 64(N): 2 A frags + 4 B frags -> 8 WMMA per k-step.
// All 6 fragment loads issued before the 8 WMMAs so the backend can cluster
// the global_load_b128s and use partial s_wait_loadcnt overlap.
// M % 32 == 0, N % 64 == 0, K % 32 == 0.

__global__ void k_gemm_bias(const unsigned short* __restrict__ A,
                            const unsigned short* __restrict__ W,
                            const float* __restrict__ bias,
                            float* __restrict__ out,
                            int M, int N, int K) {
    int wave = (blockIdx.x * blockDim.x + threadIdx.x) >> 5;
    int lane = threadIdx.x & 31;
    int tilesN = N >> 6;
    int tm = wave / tilesN;
    int tn = wave - tm * tilesN;
    if (tm >= (M >> 5)) return;             // wave-uniform exit; EXEC all-ones for WMMA
    int m0 = tm << 5, n0 = tn << 6;

    v8f acc00 = {}, acc01 = {}, acc02 = {}, acc03 = {};
    v8f acc10 = {}, acc11 = {}, acc12 = {}, acc13 = {};

    const unsigned short* A0 = A + (size_t)m0 * K;
    const unsigned short* A1 = A + (size_t)(m0 + 16) * K;
    const unsigned short* W0 = W + (size_t)(n0)      * K;
    const unsigned short* W1 = W + (size_t)(n0 + 16) * K;
    const unsigned short* W2 = W + (size_t)(n0 + 32) * K;
    const unsigned short* W3 = W + (size_t)(n0 + 48) * K;

    for (int k0 = 0; k0 < K; k0 += 32) {
        v16bf a0 = load_a_frag(A0 + k0, K, lane);
        v16bf a1 = load_a_frag(A1 + k0, K, lane);
        v16bf b0 = load_b_frag(W0 + k0, K, lane);
        v16bf b1 = load_b_frag(W1 + k0, K, lane);
        v16bf b2 = load_b_frag(W2 + k0, K, lane);
        v16bf b3 = load_b_frag(W3 + k0, K, lane);
        acc00 = wmma_bf16(a0, b0, acc00);
        acc10 = wmma_bf16(a1, b0, acc10);
        acc01 = wmma_bf16(a0, b1, acc01);
        acc11 = wmma_bf16(a1, b1, acc11);
        acc02 = wmma_bf16(a0, b2, acc02);
        acc12 = wmma_bf16(a1, b2, acc12);
        acc03 = wmma_bf16(a0, b3, acc03);
        acc13 = wmma_bf16(a1, b3, acc13);
    }

    int r = lane & 15, half = lane >> 4;
    float bb0 = bias[n0 + r];
    float bb1 = bias[n0 + 16 + r];
    float bb2 = bias[n0 + 32 + r];
    float bb3 = bias[n0 + 48 + r];
#pragma unroll
    for (int v = 0; v < 8; ++v) {
        int row0 = m0 + half * 8 + v;
        float* o0 = out + (size_t)row0 * N;
        o0[n0 + r]      = acc00[v] + bb0;
        o0[n0 + 16 + r] = acc01[v] + bb1;
        o0[n0 + 32 + r] = acc02[v] + bb2;
        o0[n0 + 48 + r] = acc03[v] + bb3;
        float* o1 = o0 + (size_t)16 * N;
        o1[n0 + r]      = acc10[v] + bb0;
        o1[n0 + 16 + r] = acc11[v] + bb1;
        o1[n0 + 32 + r] = acc12[v] + bb2;
        o1[n0 + 48 + r] = acc13[v] + bb3;
    }
}

// ---------------- Persistent fused GRU layer ----------------
// One launch runs all 256 timesteps; 64 workgroups x 8 waves = 512 waves,
// exactly the (16 batch-tiles x 32 col-tiles) tiling of the [256,512] state.
// Steps separated by a device-scope arrive-and-spin grid barrier (monotonic
// counter, re-zeroed each kernel_launch so graph replay is deterministic).
// ha is the zero-initialized ping buffer; final hidden ends in ha (T even).

__global__ void k_gru_persistent(unsigned short* __restrict__ ha,
                                 unsigned short* __restrict__ hb,
                                 const unsigned short* __restrict__ Whh,  // [1536,512] bf16
                                 const float* __restrict__ bhh,           // [1536]
                                 const float* __restrict__ xg,            // [256*256,1536] f32
                                 unsigned short* __restrict__ yout,       // [256*256,512] bf16 or null
                                 unsigned* __restrict__ barrier_ctr) {
    const int T = 256, H = 512;
    int w = blockIdx.x * 8 + (threadIdx.x >> 5);
    int lane = threadIdx.x & 31;
    int tm = w >> 5;                 // 0..15  (batch tile)
    int tn = w & 31;                 // 0..31  (hidden-col tile)
    int m0 = tm << 4, n0 = tn << 4;
    int r = lane & 15, half = lane >> 4;
    int col = n0 + r;

    const unsigned short* WR = Whh + (size_t)(n0)         * H;
    const unsigned short* WZ = Whh + (size_t)(H + n0)     * H;
    const unsigned short* WN = Whh + (size_t)(2 * H + n0) * H;
    float bR = bhh[col], bZ = bhh[H + col], bN = bhh[2 * H + col];

    unsigned target = 0;
    for (int t = 0; t < T; ++t) {
        const unsigned short* hin  = (t & 1) ? hb : ha;
        unsigned short*       hout = (t & 1) ? ha : hb;

        v8f aR = {}, aZ = {}, aN = {};
        const unsigned short* Ab = hin + (size_t)m0 * H;
        for (int k0 = 0; k0 < H; k0 += 32) {
            v16bf a   = load_a_frag(Ab + k0, H, lane);
            v16bf bRf = load_b_frag(WR + k0, H, lane);
            v16bf bZf = load_b_frag(WZ + k0, H, lane);
            v16bf bNf = load_b_frag(WN + k0, H, lane);
            aR = wmma_bf16(a, bRf, aR);
            aZ = wmma_bf16(a, bZf, aZ);
            aN = wmma_bf16(a, bNf, aN);
        }

#pragma unroll
        for (int v = 0; v < 8; ++v) {
            int b = m0 + half * 8 + v;
            const float* xrow = xg + ((size_t)b * T + t) * (3 * H);
            float xr = xrow[col], xz = xrow[H + col], xn = xrow[2 * H + col];
            float hr = aR[v] + bR, hz = aZ[v] + bZ, hn = aN[v] + bN;
            float rg = 1.0f / (1.0f + __expf(-(xr + hr)));
            float zg = 1.0f / (1.0f + __expf(-(xz + hz)));
            float ng = tanhf(xn + rg * hn);
            float hp = bf2f(hin[(size_t)b * H + col]);
            float hv = (1.0f - zg) * ng + zg * hp;
            unsigned short hbv = f2bf(hv);
            hout[(size_t)b * H + col] = hbv;
            if (yout) yout[((size_t)b * T + t) * H + col] = hbv;
        }

        // ---- grid barrier between timesteps (skip after final step) ----
        if (t != T - 1) {
            target += gridDim.x;
            __threadfence();                 // make this thread's h stores agent-visible
            __syncthreads();
            if (threadIdx.x == 0) {
                __hip_atomic_fetch_add(barrier_ctr, 1u, __ATOMIC_RELEASE,
                                       __HIP_MEMORY_SCOPE_AGENT);
                while (__hip_atomic_load(barrier_ctr, __ATOMIC_ACQUIRE,
                                         __HIP_MEMORY_SCOPE_AGENT) < target)
                    __builtin_amdgcn_s_sleep(1);
            }
            __syncthreads();
            __threadfence();                 // invalidate stale lines before reading new h
        }
    }
}

// ---------------- Head: out = relu(relu(h1) @ fc1^T + b1) @ fc2^T + b2 ----------------

__global__ void k_head(const unsigned short* __restrict__ h1,   // [256,512] bf16
                       const float* __restrict__ fc1w, const float* __restrict__ fc1b,
                       const float* __restrict__ fc2w, const float* __restrict__ fc2b,
                       float* __restrict__ out) {
    const int H = 512, F = 128, C = 10;
    __shared__ float s[F];
    int b = blockIdx.x;
    int j = threadIdx.x;                      // 128 threads
    const unsigned short* hr = h1 + (size_t)b * H;
    float acc = fc1b[j];
    const float* wrow = fc1w + (size_t)j * H;
    for (int k = 0; k < H; ++k) {
        float hv = bf2f(hr[k]);
        hv = hv > 0.0f ? hv : 0.0f;
        acc += hv * wrow[k];
    }
    s[j] = acc > 0.0f ? acc : 0.0f;
    __syncthreads();
    if (j < C) {
        float o = fc2b[j];
        const float* w2 = fc2w + (size_t)j * F;
        for (int k = 0; k < F; ++k) o += s[k] * w2[k];
        out[(size_t)b * C + j] = o;
    }
}

// ---------------- host launch ----------------

extern "C" void kernel_launch(void* const* d_in, const int* in_sizes, int n_in,
                              void* d_out, int out_size, void* d_ws, size_t ws_size,
                              hipStream_t stream) {
    (void)in_sizes; (void)n_in; (void)out_size; (void)ws_size;
    const int B = 256, T = 256, D = 128, H = 512;
    const int M = B * T;            // 65536
    const int G = 3 * H;            // 1536

    const float* x     = (const float*)d_in[0];
    const float* Wih0  = (const float*)d_in[1];
    const float* Whh0  = (const float*)d_in[2];
    const float* bih0  = (const float*)d_in[3];
    const float* bhh0  = (const float*)d_in[4];
    const float* Wih1  = (const float*)d_in[5];
    const float* Whh1  = (const float*)d_in[6];
    const float* bih1  = (const float*)d_in[7];
    const float* bhh1  = (const float*)d_in[8];
    const float* fc1w  = (const float*)d_in[9];
    const float* fc1b  = (const float*)d_in[10];
    const float* fc2w  = (const float*)d_in[11];
    const float* fc2b  = (const float*)d_in[12];
    float* out = (float*)d_out;

    // workspace carve-up
    char* p = (char*)d_ws;
    float*          xg    = (float*)p;           p += (size_t)M * G * 4;   // 402 MB (shared by both layers)
    unsigned short* y0    = (unsigned short*)p;  p += (size_t)M * H * 2;   // 67 MB
    unsigned short* xbf   = (unsigned short*)p;  p += (size_t)M * D * 2;   // 16.8 MB
    unsigned short* wih0b = (unsigned short*)p;  p += (size_t)G * D * 2;
    unsigned short* whh0b = (unsigned short*)p;  p += (size_t)G * H * 2;
    unsigned short* wih1b = (unsigned short*)p;  p += (size_t)G * H * 2;
    unsigned short* whh1b = (unsigned short*)p;  p += (size_t)G * H * 2;
    unsigned short* h0a   = (unsigned short*)p;  p += (size_t)B * H * 2;
    unsigned short* h0b   = (unsigned short*)p;  p += (size_t)B * H * 2;
    unsigned short* h1a   = (unsigned short*)p;  p += (size_t)B * H * 2;
    unsigned short* h1b   = (unsigned short*)p;  p += (size_t)B * H * 2;
    unsigned*       ctr   = (unsigned*)p;        // 2 barrier counters

    auto cvt = [&](const float* s, unsigned short* d, int n) {
        k_f32_to_bf16<<<(n + 255) / 256, 256, 0, stream>>>(s, d, n);
    };

    // init hidden-state ping buffers and barrier counters (captured in graph ->
    // deterministic across replays)
    k_zero_bf16<<<(B * H + 255) / 256, 256, 0, stream>>>(h0a, B * H);
    k_zero_bf16<<<(B * H + 255) / 256, 256, 0, stream>>>(h1a, B * H);
    k_zero_u32<<<1, 32, 0, stream>>>(ctr, 2);

    // fp32 -> bf16 conversions (weights stay resident in L2)
    cvt(x,    xbf,   M * D);
    cvt(Wih0, wih0b, G * D);
    cvt(Whh0, whh0b, G * H);
    cvt(Wih1, wih1b, G * H);
    cvt(Whh1, whh1b, G * H);

    // layer-0 input projection: xg = x @ Wih0^T + bih0  (K=128)
    {
        int waves = (M / 32) * (G / 64);       // 49152
        k_gemm_bias<<<waves / 8, 256, 0, stream>>>(xbf, wih0b, bih0, xg, M, G, D);
    }

    // layer-0 recurrence: one persistent kernel, 256 steps, grid-barrier synced
    k_gru_persistent<<<64, 256, 0, stream>>>(h0a, h0b, whh0b, bhh0, xg, y0, &ctr[0]);

    // layer-1 input projection: xg = y0 @ Wih1^T + bih1  (K=512), reuse xg buffer
    {
        int waves = (M / 32) * (G / 64);
        k_gemm_bias<<<waves / 8, 256, 0, stream>>>(y0, wih1b, bih1, xg, M, G, H);
    }

    // layer-1 recurrence, final hidden only (ends in h1a)
    k_gru_persistent<<<64, 256, 0, stream>>>(h1a, h1b, whh1b, bhh1, xg, nullptr, &ctr[1]);

    // MLP head
    k_head<<<B, 128, 0, stream>>>(h1a, fc1w, fc1b, fc2w, fc2b, out);
}
